// FractalResonanceTransformer_54228257079875
// MI455X (gfx1250) — matrix operand order repaired and checked by reading, not compiled, not run on hardware
//
#include <hip/hip_runtime.h>
#include <hip/hip_bf16.h>

// ---------------------------------------------------------------------------
// Types for CDNA5 WMMA (wave32): v_wmma_f32_16x16x32_bf16
// ---------------------------------------------------------------------------
typedef __bf16 bf16;
typedef __attribute__((ext_vector_type(16))) __bf16 v16bf;
typedef __attribute__((ext_vector_type(8)))  __bf16 v8bf;
typedef __attribute__((ext_vector_type(8)))  float  v8f;

constexpr int BB  = 2;
constexpr int SS  = 2048;
constexpr int DD  = 1024;
constexpr int HH  = 16;
constexpr int DHH = 64;

static __device__ __forceinline__ v16bf make_v16(v8bf lo, v8bf hi) {
  v16bf r;
#pragma unroll
  for (int i = 0; i < 8; ++i) { r[i] = lo[i]; r[8 + i] = hi[i]; }
  return r;
}

static __device__ __forceinline__ v8f wmma_bf16(v16bf a, v16bf b, v8f c) {
  // 8-arg form: (neg_a, A, neg_b, B, c_mod, C, reuse_a, reuse_b)
  return __builtin_amdgcn_wmma_f32_16x16x32_bf16(false, a, false, b,
                                                 (short)0, c, false, false);
}

// ---------------------------------------------------------------------------
// gfx1250 async memory->LDS DMA (no VGPR round-trip, tracked by ASYNCcnt).
// Per ISA 10.2, a flat pointer into the LDS aperture carries the LDS byte
// offset in its low 32 bits, so truncation yields a valid LDS address.
// ---------------------------------------------------------------------------
static __device__ __forceinline__ void async_ld16(void* lds_dst, const void* gsrc) {
  unsigned l = (unsigned)(unsigned long long)lds_dst;
  unsigned long long g = (unsigned long long)gsrc;
  asm volatile("global_load_async_to_lds_b128 %0, %1, off"
               :: "v"(l), "v"(g) : "memory");
}
static __device__ __forceinline__ void async_wait0() {
  asm volatile("s_wait_asynccnt 0" ::: "memory");
}

// ---------------------------------------------------------------------------
// Kernel 1: fp32 -> bf16 convert (weights)
// ---------------------------------------------------------------------------
__global__ void cvt_bf16_kernel(const float* __restrict__ src,
                                bf16* __restrict__ dst, int n) {
  int i = blockIdx.x * blockDim.x + threadIdx.x;
  if (i < n) dst[i] = (bf16)src[i];
}

// ---------------------------------------------------------------------------
// Kernel 2: LayerNorm row (1024) -> bf16 h
// ---------------------------------------------------------------------------
__global__ void ln_kernel(const float* __restrict__ x,
                          const float* __restrict__ gamma,
                          const float* __restrict__ beta,
                          bf16* __restrict__ h_bf) {
  __shared__ float red[256];
  int row = blockIdx.x, tid = threadIdx.x;
  const float* xr = x + (size_t)row * DD;
  float s = 0.f, s2 = 0.f;
  for (int i = tid; i < DD; i += 256) { float v = xr[i]; s += v; s2 += v * v; }
  red[tid] = s; __syncthreads();
  for (int st = 128; st > 0; st >>= 1) { if (tid < st) red[tid] += red[tid + st]; __syncthreads(); }
  float mu = red[0] / DD; __syncthreads();
  red[tid] = s2; __syncthreads();
  for (int st = 128; st > 0; st >>= 1) { if (tid < st) red[tid] += red[tid + st]; __syncthreads(); }
  float var = red[0] / DD - mu * mu;
  float rs  = rsqrtf(var + 1e-5f);
  for (int i = tid; i < DD; i += 256) {
    float v = (xr[i] - mu) * rs * gamma[i] + beta[i];
    h_bf[(size_t)row * DD + i] = (bf16)v;
  }
}

// ---------------------------------------------------------------------------
// Kernel 3: carrier resonance Th(b,h,s) and log(Th+1e-6)
// ---------------------------------------------------------------------------
__global__ void reson_kernel(const float* __restrict__ phase,
                             const float* __restrict__ carrier,
                             float* __restrict__ Th,
                             float* __restrict__ logTh) {
  int i = blockIdx.x * blockDim.x + threadIdx.x;  // i = (b*HH + h)*SS + s
  if (i >= BB * HH * SS) return;
  int s  = i & (SS - 1);
  int bh = i / SS;
  int h  = bh & (HH - 1);
  int b  = bh / HH;
  float t = cosf((phase[b * SS + s] - carrier[h]) * 0.5f);
  float T = t * t;
  Th[i]    = T;
  logTh[i] = logf(T + 1e-6f);
}

// ---------------------------------------------------------------------------
// Kernel 4/6: LDS-staged WMMA GEMM  C(M,N) = A(M,K) * B(K,N)
//   mode 0: store bf16
//   mode 1: per-head fractal compression (sign * |v|^exp[head]) -> bf16
//   mode 2: fp32 out = xres + acc (residual epilogue)
// Block: 128 threads = 4 waves (2x2), tile 64x64, BK=32, wave tile 32x32.
// ---------------------------------------------------------------------------
__global__ void __launch_bounds__(128)
gemm_wmma_kernel(const bf16* __restrict__ A, const bf16* __restrict__ Bm,
                 void* __restrict__ out, int M, int N, int K, int mode,
                 const float* __restrict__ exps, const float* __restrict__ xres) {
  __shared__ __align__(16) bf16 As[64][40];   // row stride 80B (16B-aligned rows)
  __shared__ __align__(16) bf16 Bt[64][40];   // B stored transposed: [n][k]
  int tid = threadIdx.x, lane = tid & 31, w = tid >> 5;
  int wr = w >> 1, wc = w & 1;
  int ln15 = lane & 15, hi = lane >> 4;
  int m0 = blockIdx.y * 64, n0 = blockIdx.x * 64;

  v8f zero = {};
  v8f acc[2][2];
  acc[0][0] = zero; acc[0][1] = zero; acc[1][0] = zero; acc[1][1] = zero;

  for (int k0 = 0; k0 < K; k0 += 32) {
    __syncthreads();
    { // stage A tile 64x32 via async DMA to LDS (two 16B chunks per thread)
      int row = tid >> 1, half = tid & 1;
      const bf16* src = &A[(size_t)(m0 + row) * K + k0 + half * 16];
      async_ld16(&As[row][half * 16],     src);
      async_ld16(&As[row][half * 16 + 8], src + 8);
    }
    // stage B tile transposed (coalesced global reads along n, ds scatter)
    for (int idx = tid; idx < 2048; idx += 128) {
      int k = idx >> 6, n = idx & 63;
      Bt[n][k] = Bm[(size_t)(k0 + k) * N + n0 + n];
    }
    async_wait0();
    __syncthreads();
    if (k0 + 32 < K)  // -> global_prefetch_b8 for next A tile
      __builtin_prefetch(&A[(size_t)(m0 + (tid >> 1)) * K + k0 + 32], 0, 0);

    // A-fragment layout (16-bit 16x32): lane m=lane&15, kbase=(lane>>4)*8,
    // elements e<8 -> K=kbase+e, e>=8 -> K=kbase+16+(e-8)
    v16bf af[2], bfv[2];
#pragma unroll
    for (int mt = 0; mt < 2; ++mt) {
      int m = wr * 32 + mt * 16 + ln15;
      int kb8 = hi * 8;
      const v8bf* p0 = (const v8bf*)&As[m][kb8];
      const v8bf* p1 = (const v8bf*)&As[m][kb8 + 16];
      af[mt] = make_v16(p0[0], p1[0]);
    }
    // B-fragment layout (16-bit 32x16): lane n=lane&15, kbase=(lane>>4)*8,
    // elements e<8 -> K=kbase+e, e>=8 -> K=kbase+16+(e-8)
#pragma unroll
    for (int nt = 0; nt < 2; ++nt) {
      int n = wc * 32 + nt * 16 + ln15;
      int kb16 = hi * 8;
      const v8bf* p0 = (const v8bf*)&Bt[n][kb16];
      const v8bf* p1 = (const v8bf*)&Bt[n][kb16 + 16];
      bfv[nt] = make_v16(p0[0], p1[0]);
    }
#pragma unroll
    for (int mt = 0; mt < 2; ++mt)
#pragma unroll
      for (int nt = 0; nt < 2; ++nt)
        acc[mt][nt] = wmma_bf16(af[mt], bfv[nt], acc[mt][nt]);
  }

  // C/D layout: VGPR r -> M = r + 8*(lane>>4), N = lane&15
#pragma unroll
  for (int mt = 0; mt < 2; ++mt) {
#pragma unroll
    for (int nt = 0; nt < 2; ++nt) {
#pragma unroll
      for (int r = 0; r < 8; ++r) {
        int gm = m0 + wr * 32 + mt * 16 + hi * 8 + r;
        int gn = n0 + wc * 32 + nt * 16 + ln15;
        float v = acc[mt][nt][r];
        if (mode == 1) {
          float e = exps[gn >> 6];  // head = column / DH
          float sgn = (v < 0.0f) ? -1.0f : 1.0f;
          v = sgn * __powf(fmaxf(fabsf(v), 1e-8f), e);
        }
        size_t off = (size_t)gm * N + gn;
        if (mode == 2) ((float*)out)[off] = xres[off] + v;
        else           ((bf16*)out)[off]  = (bf16)v;
      }
    }
  }
}

// ---------------------------------------------------------------------------
// Kernel 5: flash attention (causal) with resonance bias/gating.
// Block = 64 query rows (4 waves x 16 rows), grid (S/64, H, B).
// ---------------------------------------------------------------------------
__global__ void __launch_bounds__(128)
attn_kernel(const bf16* __restrict__ q_bf, const bf16* __restrict__ k_bf,
            const bf16* __restrict__ v_bf, const float* __restrict__ Th,
            const float* __restrict__ logTh, const float* __restrict__ balance,
            bf16* __restrict__ o_bf) {
  __shared__ __align__(16) bf16 Ks[64][72];     // key tile, row-major (key, dh)
  __shared__ __align__(16) bf16 Vt[64][72];     // value tile transposed (dh, key)
  __shared__ __align__(16) bf16 Pl[4][16][72];  // per-wave P staging
  int tid = threadIdx.x, lane = tid & 31, wave = tid >> 5;
  int ln15 = lane & 15, hi = lane >> 4;
  int qt = blockIdx.x, h = blockIdx.y, b = blockIdx.z;
  int qbase = qt * 64;
  int mbase = qbase + wave * 16 + hi * 8;       // this lane's first M row
  size_t bh_off = (size_t)(b * HH + h) * SS;

  float bal = balance[0];
  float tau = 1.0f / (2.0f * bal + 1e-8f);
  tau = fminf(fmaxf(tau, 0.1f), 10.0f);
  float inv_tau = 1.0f / tau;
  const float inv_sqrt_dh = 0.125f;             // 1/sqrt(64)

  // Q fragments (kept in VGPRs for the whole K loop)
  v16bf qfrag[2];
  {
    int qrow = qbase + wave * 16 + ln15;
    const bf16* qp = q_bf + (size_t)(b * SS + qrow) * DD + h * DHH;
    int kb8 = hi * 8;
#pragma unroll
    for (int kc = 0; kc < 2; ++kc) {
      const v8bf* p0 = (const v8bf*)(qp + kc * 32 + kb8);
      const v8bf* p1 = (const v8bf*)(qp + kc * 32 + kb8 + 16);
      qfrag[kc] = make_v16(p0[0], p1[0]);
    }
  }

  v8f zero = {};
  v8f oacc[4]; oacc[0] = zero; oacc[1] = zero; oacc[2] = zero; oacc[3] = zero;
  float m_i[8], l_i[8];
#pragma unroll
  for (int r = 0; r < 8; ++r) { m_i[r] = -3.0e38f; l_i[r] = 0.0f; }

  for (int jt = 0; jt <= qt; ++jt) {            // causal: only tiles <= qt
    int kb = jt * 64;
    __syncthreads();
    { // stage K tile via async DMA to LDS (four 16B chunks per thread)
      int row = tid >> 1, half = tid & 1;
      const bf16* src = k_bf + (size_t)(b * SS + kb + row) * DD + h * DHH + half * 32;
#pragma unroll
      for (int c = 0; c < 4; ++c)
        async_ld16(&Ks[row][half * 32 + c * 8], src + c * 8);
    }
    // stage V transposed: Vt[dh][key]
    for (int idx = tid; idx < 4096; idx += 128) {
      int key = idx >> 6, dh = idx & 63;
      Vt[dh][key] = v_bf[(size_t)(b * SS + kb + key) * DD + h * DHH + dh];
    }
    async_wait0();
    __syncthreads();

    // S = Q * K^T : 4 subtiles of 16 keys, K-dim = dh (2 chunks of 32)
    v8f sc[4]; sc[0] = zero; sc[1] = zero; sc[2] = zero; sc[3] = zero;
#pragma unroll
    for (int kc = 0; kc < 2; ++kc) {
#pragma unroll
      for (int nt = 0; nt < 4; ++nt) {
        int krow = nt * 16 + ln15;              // key index in tile (WMMA N)
        int kb16 = hi * 8;
        const v8bf* p0 = (const v8bf*)&Ks[krow][kc * 32 + kb16];
        const v8bf* p1 = (const v8bf*)&Ks[krow][kc * 32 + kb16 + 16];
        sc[nt] = wmma_bf16(qfrag[kc], make_v16(p0[0], p1[0]), sc[nt]);
      }
    }

    // logits = (qk*scale + logTh_key)/tau, causal mask, online softmax
    float rowmax[8];
#pragma unroll
    for (int r = 0; r < 8; ++r) rowmax[r] = -3.0e38f;
#pragma unroll
    for (int nt = 0; nt < 4; ++nt) {
      int n_g = kb + nt * 16 + ln15;
      float lt = logTh[bh_off + n_g];
#pragma unroll
      for (int r = 0; r < 8; ++r) {
        float v = (sc[nt][r] * inv_sqrt_dh + lt) * inv_tau;
        if (n_g > mbase + r) v = -3.0e38f;      // only bites on diagonal tile
        sc[nt][r] = v;
        rowmax[r] = fmaxf(rowmax[r], v);
      }
    }
    float alpha[8];
#pragma unroll
    for (int r = 0; r < 8; ++r) {               // reduce across 16-lane half
      float m = rowmax[r];
      m = fmaxf(m, __shfl_xor(m, 1));
      m = fmaxf(m, __shfl_xor(m, 2));
      m = fmaxf(m, __shfl_xor(m, 4));
      m = fmaxf(m, __shfl_xor(m, 8));
      float mnew = fmaxf(m_i[r], m);
      alpha[r] = __expf(m_i[r] - mnew);
      m_i[r] = mnew;
      l_i[r] *= alpha[r];
    }
    float rowsum[8];
#pragma unroll
    for (int r = 0; r < 8; ++r) rowsum[r] = 0.0f;
#pragma unroll
    for (int nt = 0; nt < 4; ++nt)
#pragma unroll
      for (int r = 0; r < 8; ++r) {
        float p = __expf(sc[nt][r] - m_i[r]);
        sc[nt][r] = p;
        rowsum[r] += p;
      }
#pragma unroll
    for (int r = 0; r < 8; ++r) {
      float s = rowsum[r];
      s += __shfl_xor(s, 1); s += __shfl_xor(s, 2);
      s += __shfl_xor(s, 4); s += __shfl_xor(s, 8);
      l_i[r] += s;
#pragma unroll
      for (int nt = 0; nt < 4; ++nt) oacc[nt][r] *= alpha[r];
    }

    // P (D-layout) -> per-wave LDS -> A-layout fragments for P*V
#pragma unroll
    for (int nt = 0; nt < 4; ++nt)
#pragma unroll
      for (int r = 0; r < 8; ++r)
        Pl[wave][hi * 8 + r][nt * 16 + ln15] = (bf16)sc[nt][r];
    asm volatile("s_wait_dscnt 0" ::: "memory");  // wave-local LDS RAW fence

#pragma unroll
    for (int kc = 0; kc < 2; ++kc) {
      int kb8 = hi * 8;
      const v8bf* a0 = (const v8bf*)&Pl[wave][ln15][kc * 32 + kb8];
      const v8bf* a1 = (const v8bf*)&Pl[wave][ln15][kc * 32 + kb8 + 16];
      v16bf af = make_v16(a0[0], a1[0]);
#pragma unroll
      for (int nt = 0; nt < 4; ++nt) {
        int nrow = nt * 16 + ln15;              // dh column (WMMA N)
        int kb16 = hi * 8;
        const v8bf* b0 = (const v8bf*)&Vt[nrow][kc * 32 + kb16];
        const v8bf* b1 = (const v8bf*)&Vt[nrow][kc * 32 + kb16 + 16];
        oacc[nt] = wmma_bf16(af, make_v16(b0[0], b1[0]), oacc[nt]);
      }
    }
  }

  // epilogue: O = (O / l) * Th_query, store bf16 in (B,S,D) layout
#pragma unroll
  for (int r = 0; r < 8; ++r) {
    int mg = mbase + r;
    float invl = 1.0f / l_i[r];
    float thq = Th[bh_off + mg];
#pragma unroll
    for (int nt = 0; nt < 4; ++nt) {
      int col = h * DHH + nt * 16 + ln15;
      o_bf[(size_t)(b * SS + mg) * DD + col] = (bf16)(oacc[nt][r] * invl * thq);
    }
  }
}

// ---------------------------------------------------------------------------
// Launch
// ---------------------------------------------------------------------------
extern "C" void kernel_launch(void* const* d_in, const int* in_sizes, int n_in,
                              void* d_out, int out_size, void* d_ws, size_t ws_size,
                              hipStream_t stream) {
  (void)in_sizes; (void)n_in; (void)out_size; (void)ws_size;
  const float* x       = (const float*)d_in[0];
  const float* phase   = (const float*)d_in[1];
  const float* Wq      = (const float*)d_in[2];
  const float* Wk      = (const float*)d_in[3];
  const float* Wv      = (const float*)d_in[4];
  const float* Wo      = (const float*)d_in[5];
  const float* gamma   = (const float*)d_in[6];
  const float* beta    = (const float*)d_in[7];
  const float* expo    = (const float*)d_in[8];
  const float* carrier = (const float*)d_in[9];
  const float* balance = (const float*)d_in[10];
  float* out = (float*)d_out;

  const int M = BB * SS;                 // 4096
  const size_t SZ_HD = (size_t)M * DD * sizeof(bf16);   // 8 MB
  const size_t SZ_W  = (size_t)DD * DD * sizeof(bf16);  // 2 MB
  const size_t SZ_TH = (size_t)BB * HH * SS * sizeof(float);

  char* w = (char*)d_ws;
  bf16* h_bf   = (bf16*)w; w += SZ_HD;
  bf16* q_bf   = (bf16*)w; w += SZ_HD;
  bf16* k_bf   = (bf16*)w; w += SZ_HD;
  bf16* v_bf   = (bf16*)w; w += SZ_HD;
  bf16* o_bf   = (bf16*)w; w += SZ_HD;
  bf16* wq_bf  = (bf16*)w; w += SZ_W;
  bf16* wk_bf  = (bf16*)w; w += SZ_W;
  bf16* wv_bf  = (bf16*)w; w += SZ_W;
  bf16* wo_bf  = (bf16*)w; w += SZ_W;
  float* ThA   = (float*)w; w += SZ_TH;
  float* logThA= (float*)w; w += SZ_TH;

  const int NW = DD * DD;
  cvt_bf16_kernel<<<(NW + 255) / 256, 256, 0, stream>>>(Wq, wq_bf, NW);
  cvt_bf16_kernel<<<(NW + 255) / 256, 256, 0, stream>>>(Wk, wk_bf, NW);
  cvt_bf16_kernel<<<(NW + 255) / 256, 256, 0, stream>>>(Wv, wv_bf, NW);
  cvt_bf16_kernel<<<(NW + 255) / 256, 256, 0, stream>>>(Wo, wo_bf, NW);

  ln_kernel<<<M, 256, 0, stream>>>(x, gamma, beta, h_bf);

  const int NT = BB * HH * SS;
  reson_kernel<<<(NT + 255) / 256, 256, 0, stream>>>(phase, carrier, ThA, logThA);

  dim3 gg(DD / 64, M / 64);              // (16, 64)
  gemm_wmma_kernel<<<gg, 128, 0, stream>>>(h_bf, wq_bf, q_bf, M, DD, DD, 0, nullptr, nullptr);
  gemm_wmma_kernel<<<gg, 128, 0, stream>>>(h_bf, wk_bf, k_bf, M, DD, DD, 1, expo, nullptr);
  gemm_wmma_kernel<<<gg, 128, 0, stream>>>(h_bf, wv_bf, v_bf, M, DD, DD, 1, expo, nullptr);

  attn_kernel<<<dim3(SS / 64, HH, BB), 128, 0, stream>>>(q_bf, k_bf, v_bf, ThA,
                                                         logThA, balance, o_bf);

  gemm_wmma_kernel<<<gg, 128, 0, stream>>>(o_bf, wo_bf, (void*)out, M, DD, DD, 2,
                                           nullptr, x);
}
